// EntropicNeuralQuantileRegression_16707422781545
// MI455X (gfx1250) — matrix operand order repaired and checked by reading, not compiled, not run on hardware
//
#include <hip/hip_runtime.h>
#include <math.h>

typedef __attribute__((ext_vector_type(16))) _Float16 v16h;
typedef __attribute__((ext_vector_type(8)))  _Float16 v8h;
typedef __attribute__((ext_vector_type(4)))  _Float16 v4h;
typedef __attribute__((ext_vector_type(8)))  float    v8f;

#define FDIM 64
#define RDIM 8
#define HDIM 128
#define EPSV 0.1f
#define NROWS 1024
#define MCOLS 1024
#define JCHUNK 128
#define NCHUNKS 8      // MCOLS / JCHUNK
#define IBLK 16
#define JSUB 16
#define PAIRS 256      // IBLK * JSUB

#define LOG2E 1.4426950408889634f
#define LN2   0.6931471805599453f

// Raw hardware transcendentals (base-2 semantics): one v_exp_f32 / v_log_f32.
__device__ __forceinline__ float fast_exp2(float x) { return __builtin_amdgcn_exp2f(x); }
__device__ __forceinline__ float fast_log2(float x) { return __builtin_amdgcn_logf(x); }

// ---------------------------------------------------------------------------
// Prep: A = X @ W0[:64,:]  (f32, 1024x128)
//       Bu = U @ W0[64:,:] + b0  (f32, 1024x128)
// ---------------------------------------------------------------------------
__global__ __launch_bounds__(128) void prep_ab_kernel(
    const float* __restrict__ X, const float* __restrict__ U,
    const float* __restrict__ W0, const float* __restrict__ b0,
    float* __restrict__ A, float* __restrict__ Bu) {
  int row = blockIdx.x;     // 0..2047
  int k = threadIdx.x;      // 0..127
  if (row < NROWS) {
    const float* x = X + row * FDIM;
    float acc = 0.0f;
#pragma unroll
    for (int f = 0; f < FDIM; ++f) acc = fmaf(x[f], W0[f * HDIM + k], acc);
    A[row * HDIM + k] = acc;
  } else {
    int j = row - NROWS;
    const float* u = U + j * RDIM;
    float acc = b0[k];
#pragma unroll
    for (int r = 0; r < RDIM; ++r) acc = fmaf(u[r], W0[(FDIM + r) * HDIM + k], acc);
    Bu[j * HDIM + k] = acc;
  }
}

// ---------------------------------------------------------------------------
// Fast stable softplus using the native v_exp_f32/v_log_f32 pipes:
//   softplus(x) = max(x,0) + ln2 * log2(1 + exp2(-|x|*log2e))
// ~6 VALU/TRANS ops instead of libm log1pf's ~20.
__device__ __forceinline__ float softplusf(float x) {
  float e = fast_exp2(-fabsf(x) * LOG2E);     // exp(-|x|): 1 mul + v_exp_f32
  float l = fast_log2(1.0f + e);              // 1 add + v_log_f32
  return fmaf(l, LN2, fmaxf(x, 0.0f));        // 1 fma + 1 max
}

// A-fragment (16x32 f16, M across lanes): lane l holds row M=l&15;
// K = (l>=16 ? 8:0) + (e&7) + (e>=8 ? 16:0) + 32*kb.  Two b128 loads.
__device__ __forceinline__ v16h load_a_frag(const _Float16* sH, int m0, int kb, int lane) {
  int row = m0 + (lane & 15);
  int base = (lane >= 16) ? 8 : 0;
  const _Float16* p = sH + row * HDIM + kb * 32 + base;
  v8h lo = *(const v8h*)p;
  v8h hi = *(const v8h*)(p + 16);
  return __builtin_shufflevector(lo, hi, 0, 1, 2, 3, 4, 5, 6, 7,
                                 8, 9, 10, 11, 12, 13, 14, 15);
}

// B-fragment (32x16 f16, N across lanes): lane l holds column N=l&15,
// K = (l>=16 ? 8:0) + (e&7) + (e>=8 ? 16:0) + 32*kb.  sW stored transposed [N][K].
__device__ __forceinline__ v16h load_b_frag(const _Float16* sW, int n, int kb, int lane) {
  int col = n * 16 + (lane & 15);
  int koff = kb * 32 + ((lane >= 16) ? 8 : 0);
  const _Float16* p = sW + col * HDIM + koff;
  v8h lo = *(const v8h*)p;
  v8h hi = *(const v8h*)(p + 16);
  return __builtin_shufflevector(lo, hi, 0, 1, 2, 3, 4, 5, 6, 7,
                                 8, 9, 10, 11, 12, 13, 14, 15);
}

// One dense layer for this wave's 32-pair (2 M-tile) block:
//   sOut = softplus(sIn @ W + b)
// A-fragments hoisted across the N loop; each B-fragment feeds 2 WMMAs.
__device__ __forceinline__ void mlp_layer(const _Float16* __restrict__ sIn,
                                          _Float16* __restrict__ sOut,
                                          const _Float16* __restrict__ sW,
                                          const float* __restrict__ sBias,
                                          int wave, int lane) {
  const int m0 = wave * 32;
  const int nl = lane & 15;
  const int mhalf = (lane >= 16) ? 8 : 0;

  v16h afrag[2][4];
#pragma unroll
  for (int mt = 0; mt < 2; ++mt)
#pragma unroll
    for (int kb = 0; kb < 4; ++kb)
      afrag[mt][kb] = load_a_frag(sIn, m0 + mt * 16, kb, lane);

#pragma unroll
  for (int n = 0; n < 8; ++n) {
    float bias = sBias[n * 16 + nl];
    v8f c0 = {bias, bias, bias, bias, bias, bias, bias, bias};
    v8f c1 = c0;
#pragma unroll
    for (int kb = 0; kb < 4; ++kb) {
      v16h b = load_b_frag(sW, n, kb, lane);
      c0 = __builtin_amdgcn_wmma_f32_16x16x32_f16(false, afrag[0][kb], false, b,
                                                  (short)0, c0, false, false);
      c1 = __builtin_amdgcn_wmma_f32_16x16x32_f16(false, afrag[1][kb], false, b,
                                                  (short)0, c1, false, false);
    }
    // D layout: lane l holds N=l&15; element r -> M = mbase + r
    int N = n * 16 + nl;
    int mb0 = m0 + mhalf;
    int mb1 = m0 + 16 + mhalf;
#pragma unroll
    for (int r = 0; r < 8; ++r) {
      sOut[(mb0 + r) * HDIM + N] = (_Float16)softplusf(c0[r]);
      sOut[(mb1 + r) * HDIM + N] = (_Float16)softplusf(c1[r]);
    }
  }
}

__global__ __launch_bounds__(256) void enqr_main_kernel(
    const float* __restrict__ A, const float* __restrict__ Bu,
    const float* __restrict__ Y, const float* __restrict__ U,
    const float* __restrict__ W1, const float* __restrict__ b1,
    const float* __restrict__ W2, const float* __restrict__ b2,
    const float* __restrict__ Wout, const float* __restrict__ bout,
    float2* __restrict__ partials) {
  __shared__ _Float16 sW1[HDIM * HDIM];   // transposed [N][K], f16
  __shared__ _Float16 sW2[HDIM * HDIM];   // transposed [N][K], f16
  __shared__ _Float16 sH0[PAIRS * HDIM];  // activation ping (64 KB)
  __shared__ _Float16 sH1[PAIRS * HDIM];  // activation pong (64 KB)
  __shared__ float sA[IBLK * HDIM];       // f32 A rows for this i-block
  __shared__ float sBu[JSUB * HDIM];      // f32 Bu rows for this j-subtile
  __shared__ float sB1[HDIM], sB2[HDIM], sWo[HDIM];

  const int t = threadIdx.x;
  const int wave = t >> 5, lane = t & 31;
  const int i0 = blockIdx.x * IBLK;
  const int jc = blockIdx.y;
  const int jbase = jc * JCHUNK;

  // Stage weights (convert + transpose) and the A rows for this i-block.
  for (int idx = t; idx < HDIM * HDIM; idx += 256) {
    int n = idx >> 7, k = idx & (HDIM - 1);
    sW1[idx] = (_Float16)W1[k * HDIM + n];
    sW2[idx] = (_Float16)W2[k * HDIM + n];
  }
  for (int idx = t; idx < IBLK * HDIM; idx += 256) {
    sA[idx] = A[i0 * HDIM + idx];
  }
  if (t < HDIM) {
    sB1[t] = b1[t];
    sB2[t] = b2[t];
    sWo[t] = Wout[t];
  }
  __syncthreads();

  const float bo = bout[0];
  float m_run = -INFINITY;
  float s_run = 0.0f;

  for (int sub = 0; sub < JCHUNK / JSUB; ++sub) {
    const int j0 = jbase + sub * JSUB;

    // Stage Bu rows for this j-subtile (read each global row exactly once).
    for (int idx = t; idx < JSUB * HDIM; idx += 256) {
      sBu[idx] = Bu[j0 * HDIM + idx];
    }
    __syncthreads();

    // h0 = softplus(A[i] + Bu[j]) -> sH0 (f16), vectorized 4-wide:
    // quad q -> pair p = q>>5, k = (q&31)*4
    for (int q = t; q < PAIRS * (HDIM / 4); q += 256) {
      int p = q >> 5;
      int k = (q & 31) * 4;
      int il = p >> 4;
      int jl = p & 15;
      const float4 a4 = *(const float4*)(sA + il * HDIM + k);
      const float4 b4 = *(const float4*)(sBu + jl * HDIM + k);
      v4h h4;
      h4[0] = (_Float16)softplusf(a4.x + b4.x);
      h4[1] = (_Float16)softplusf(a4.y + b4.y);
      h4[2] = (_Float16)softplusf(a4.z + b4.z);
      h4[3] = (_Float16)softplusf(a4.w + b4.w);
      *(v4h*)(sH0 + p * HDIM + k) = h4;
    }
    __syncthreads();

    mlp_layer(sH0, sH1, sW1, sB1, wave, lane);  // layer 1
    __syncthreads();
    mlp_layer(sH1, sH0, sW2, sB2, wave, lane);  // layer 2 (h2 -> sH0)
    __syncthreads();

    // Output head + per-row online logsumexp over this j-subtile.
    {
      const v8h* hv = (const v8h*)(sH0 + t * HDIM);
      float dot = 0.0f;
#pragma unroll
      for (int kk = 0; kk < HDIM / 8; ++kk) {
        v8h h8 = hv[kk];
#pragma unroll
        for (int r = 0; r < 8; ++r) dot = fmaf((float)h8[r], sWo[kk * 8 + r], dot);
      }
      float phi = dot + bo;

      int i = i0 + (t >> 4);
      int j = j0 + (t & 15);
      float cost = 0.0f;
#pragma unroll
      for (int r = 0; r < RDIM; ++r) cost = fmaf(Y[i * RDIM + r], U[j * RDIM + r], cost);
      float slack = (cost - phi) * (1.0f / EPSV);

      // reduce over the 16 lanes holding the same i (aligned half-waves)
      float gm = slack;
#pragma unroll
      for (int off = 8; off >= 1; off >>= 1)
        gm = fmaxf(gm, __shfl_xor(gm, off, 16));
      float e = fast_exp2((slack - gm) * LOG2E);
#pragma unroll
      for (int off = 8; off >= 1; off >>= 1)
        e += __shfl_xor(e, off, 16);

      if ((t & 15) == 0) {
        float nm = fmaxf(m_run, gm);
        s_run = s_run * fast_exp2((m_run - nm) * LOG2E)
              + e    * fast_exp2((gm   - nm) * LOG2E);
        m_run = nm;
      }
    }
    __syncthreads();
  }

  if ((t & 15) == 0) {
    int i = i0 + (t >> 4);
    partials[jc * NROWS + i] = make_float2(m_run, s_run);
  }
}

// ---------------------------------------------------------------------------
// Final reduce over j-chunks: psi[i] = eps*(m + log(s) - log(MCOLS))
// ---------------------------------------------------------------------------
__global__ __launch_bounds__(256) void enqr_reduce_kernel(
    const float2* __restrict__ partials, float* __restrict__ out) {
  int i = blockIdx.x * blockDim.x + threadIdx.x;
  if (i >= NROWS) return;
  float m = -INFINITY, s = 0.0f;
#pragma unroll
  for (int c = 0; c < NCHUNKS; ++c) {
    float2 p = partials[c * NROWS + i];
    float nm = fmaxf(m, p.x);
    s = s * fast_exp2((m - nm) * LOG2E) + p.y * fast_exp2((p.x - nm) * LOG2E);
    m = nm;
  }
  out[i] = EPSV * (m + fast_log2(s) * LN2 - logf((float)MCOLS));
}

// ---------------------------------------------------------------------------
extern "C" void kernel_launch(void* const* d_in, const int* in_sizes, int n_in,
                              void* d_out, int out_size, void* d_ws, size_t ws_size,
                              hipStream_t stream) {
  const float* X    = (const float*)d_in[0];
  const float* Y    = (const float*)d_in[1];
  const float* U    = (const float*)d_in[2];
  const float* W0   = (const float*)d_in[3];
  const float* b0   = (const float*)d_in[4];
  const float* W1   = (const float*)d_in[5];
  const float* b1   = (const float*)d_in[6];
  const float* W2   = (const float*)d_in[7];
  const float* b2   = (const float*)d_in[8];
  const float* Wout = (const float*)d_in[9];
  const float* bout = (const float*)d_in[10];

  float* A  = (float*)d_ws;                        // 1024*128 f32
  float* Bu = A + NROWS * HDIM;                    // 1024*128 f32
  float2* partials = (float2*)(Bu + MCOLS * HDIM); // NCHUNKS*1024 float2

  prep_ab_kernel<<<NROWS + MCOLS, HDIM, 0, stream>>>(X, U, W0, b0, A, Bu);

  dim3 grid(NROWS / IBLK, NCHUNKS);  // 64 x 8 = 512 workgroups
  enqr_main_kernel<<<grid, 256, 0, stream>>>(A, Bu, Y, U, W1, b1, W2, b2,
                                             Wout, bout, partials);

  enqr_reduce_kernel<<<(NROWS + 255) / 256, 256, 0, stream>>>(
      partials, (float*)d_out);
}
